// StyleConv2D_72868415144290
// MI455X (gfx1250) — matrix operand order repaired and checked by reading, not compiled, not run on hardware
//
#include <hip/hip_runtime.h>

typedef __attribute__((ext_vector_type(16))) __bf16 v16bf;
typedef __attribute__((ext_vector_type(8)))  __bf16 v8bf;
typedef __attribute__((ext_vector_type(8)))  float  v8f;

#define CIN   512
#define COUT  512
#define HH    64
#define WW    64
#define KDIM  4608   // CIN * 9
#define NB    8
#define NCOLS 66     // 64 pixels + 1 halo each side
#define CST   40     // padded ci stride (80B rows, 16B aligned)

__device__ __forceinline__ unsigned short f2bf(float f) {
    union { float f; unsigned u; } v; v.f = f;
    unsigned u = v.u;
    u += 0x7FFFu + ((u >> 16) & 1u);   // round-to-nearest-even
    return (unsigned short)(u >> 16);
}

// ---------------- Kernel 1: s[b][ci] = style[b,:] . mod_w[ci,:] + mod_b[ci]
__global__ void k_modulate(const float* __restrict__ style,
                           const float* __restrict__ mod_w,
                           const float* __restrict__ mod_b,
                           float* __restrict__ sbuf) {
    const int wid  = threadIdx.x >> 5;
    const int lane = threadIdx.x & 31;
    const int wg   = blockIdx.x * (blockDim.x >> 5) + wid;  // (b, ci)
    const int b  = wg >> 9;
    const int ci = wg & (CIN - 1);
    const float* st = style + b * 512;
    const float* mw = mod_w + ci * 512;
    float acc = 0.f;
    for (int k = lane; k < 512; k += 32) acc += st[k] * mw[k];
    #pragma unroll
    for (int off = 16; off > 0; off >>= 1)
        acc += __shfl_xor(acc, off, 32);
    if (lane == 0) sbuf[wg] = acc + mod_b[ci];
}

// ---------------- Kernel 2: per (b,co) wave: demod factor + bf16 pack
// wmod layout: [b][co][(kh*3+kw)*512 + ci]  (K-major for implicit GEMM)
__global__ void k_demod_pack(const float* __restrict__ weight,
                             const float* __restrict__ sbuf,
                             unsigned short* __restrict__ wmod) {
    const float scale = 0.014731391274719738f;  // 1/sqrt(512*9)
    const int wid  = threadIdx.x >> 5;
    const int lane = threadIdx.x & 31;
    const int wg   = blockIdx.x * (blockDim.x >> 5) + wid;
    const int b  = wg >> 9;
    const int co = wg & (COUT - 1);
    const float* sb   = sbuf + b * CIN;
    const float* wrow = weight + (size_t)co * (CIN * 9);

    float ssum = 0.f;
    for (int t = lane; t < KDIM; t += 32) {
        int ci = t & (CIN - 1), khw = t >> 9;
        float m = scale * wrow[ci * 9 + khw] * sb[ci];
        ssum += m * m;
    }
    #pragma unroll
    for (int off = 16; off > 0; off >>= 1)
        ssum += __shfl_xor(ssum, off, 32);
    const float d = sqrtf(ssum + 1e-7f);   // ref divides by rsqrt -> mul by sqrt

    unsigned short* orow = wmod + (size_t)wg * KDIM;
    for (int t = lane; t < KDIM; t += 32) {
        int ci = t & (CIN - 1), khw = t >> 9;
        float m = scale * wrow[ci * 9 + khw] * sb[ci];
        orow[t] = f2bf(m * d);
    }
}

// ---------------- Kernel 3: implicit-GEMM conv, halo-staged LDS, SW-pipelined
// block = 512 thr (16 waves). Tile: 256 Cout x 128 pixels (2 full rows).
// Per ci-chunk: stage 32ci x 4rows x 66cols; next chunk's x loads are issued
// before the 72-WMMA burst so HBM latency hides behind the matrix pipe.
__global__ void __launch_bounds__(512)
k_conv_wmma(const float* __restrict__ x,
            const unsigned short* __restrict__ wmod,
            float* __restrict__ out) {
    __shared__ __align__(16) unsigned short Bsh[4 * NCOLS * CST]; // 21120 B

    const int tid  = threadIdx.x;
    const int lane = tid & 31;
    const int wid  = tid >> 5;         // 0..15
    const int blk  = blockIdx.x;
    const int h0   = (blk & 31) * 2;   // first of 2 output rows
    const int mg   = (blk >> 5) & 1;   // 2 groups of 256 Cout
    const int b    = blk >> 6;         // batch

    const int laneM = lane & 15;
    const int hi16  = lane >> 4;       // 0: lanes 0-15, 1: lanes 16-31

    // A row owned by this lane (ISA 7.12.2 16-bit A 16x32 layout)
    const size_t arow = (size_t)(b * COUT + mg * 256 + wid * 16 + laneM) * KDIM;
    const int aoff0 = hi16 ? 8 : 0;    // lanes<16: K{0..7,16..23}; else {8..15,24..31}
    const int boff  = hi16 ? 8 : 0;    // B: lanes<16 K{0..7,16..23}; else {8..15,24..31}

    v8f acc[2][4] = {{{}, {}, {}, {}}, {{}, {}, {}, {}}};

    // register staging buffer: 4 rows x up-to-5 elements per thread
    float vstage[4][5];

    // issue global loads of x chunk `cc` into vstage (no LDS traffic here)
    auto load_chunk = [&](int cc) {
        #pragma unroll
        for (int r = 0; r < 4; ++r) {
            const int hh = h0 + r - 1;
            const bool rowOK = (hh >= 0) && (hh < HH);
            #pragma unroll
            for (int j = 0; j < 5; ++j) {
                const int e = j * 512 + tid;
                float v = 0.f;
                if (e < 32 * NCOLS) {
                    const int ci   = e / NCOLS;
                    const int wimg = (e - ci * NCOLS) - 1;
                    if (rowOK && wimg >= 0 && wimg < WW)
                        v = x[((size_t)(b * CIN + cc + ci) * HH + hh) * WW + wimg];
                }
                vstage[r][j] = v;
            }
        }
    };

    load_chunk(0);   // prologue

    for (int cc = 0; cc < CIN; cc += 32) {
        __syncthreads();      // previous chunk's WMMA reads of Bsh are done
        // ---- commit staged registers to LDS as bf16, layout [row][col][ci]
        #pragma unroll
        for (int r = 0; r < 4; ++r) {
            #pragma unroll
            for (int j = 0; j < 5; ++j) {
                const int e = j * 512 + tid;
                if (e < 32 * NCOLS) {
                    const int ci  = e / NCOLS;
                    const int col = e - ci * NCOLS;
                    Bsh[(r * NCOLS + col) * CST + ci] = f2bf(vstage[r][j]);
                }
            }
        }
        __syncthreads();      // staging visible to all waves

        // ---- prefetch next chunk's x behind the WMMA burst
        if (cc + 32 < CIN) load_chunk(cc + 32);

        // ---- 9 filter shifts x 2 output rows x 4 col-subtiles from LDS
        #pragma unroll
        for (int khw = 0; khw < 9; ++khw) {
            const int dr = khw / 3;        // image-row offset 0..2
            const int dc = khw % 3;        // col shift 0..2 (== dw+1)

            // A fragment: two global_load_b128 from L2-resident wmod
            const unsigned short* ap = wmod + arow + khw * CIN + cc;
            v8bf alo = *reinterpret_cast<const v8bf*>(ap + aoff0);
            v8bf ahi = *reinterpret_cast<const v8bf*>(ap + aoff0 + 16);
            v16bf afrag;
            #pragma unroll
            for (int i = 0; i < 8; ++i) { afrag[i] = alo[i]; afrag[i + 8] = ahi[i]; }

            #pragma unroll
            for (int o = 0; o < 2; ++o) {          // output row h0+o
                const int lr = dr + o;             // LDS row 0..3
                #pragma unroll
                for (int s = 0; s < 4; ++s) {
                    const int col = s * 16 + laneM + dc;     // 0..65
                    const unsigned short* bp = &Bsh[(lr * NCOLS + col) * CST + boff];
                    v8bf blo = *reinterpret_cast<const v8bf*>(bp);
                    v8bf bhi = *reinterpret_cast<const v8bf*>(bp + 16);
                    v16bf bfrag;
                    #pragma unroll
                    for (int i = 0; i < 8; ++i) { bfrag[i] = blo[i]; bfrag[i + 8] = bhi[i]; }

                    acc[o][s] = __builtin_amdgcn_wmma_f32_16x16x32_bf16(
                                    false, afrag, false, bfrag,
                                    (short)0, acc[o][s], false, false);
                }
            }
        }
    }

    // C/D layout (7.12.2): VGPR r -> M = r + 8*hi16 ; N = laneM (+16*s)
    const int coBase = mg * 256 + wid * 16 + (hi16 ? 8 : 0);
    #pragma unroll
    for (int r = 0; r < 8; ++r) {
        const int co = coBase + r;
        #pragma unroll
        for (int o = 0; o < 2; ++o) {
            float* orow = out + (((size_t)(b * COUT + co) * HH) + h0 + o) * WW;
            #pragma unroll
            for (int s = 0; s < 4; ++s)
                orow[s * 16 + laneM] = acc[o][s][r];
        }
    }
}

extern "C" void kernel_launch(void* const* d_in, const int* in_sizes, int n_in,
                              void* d_out, int out_size, void* d_ws, size_t ws_size,
                              hipStream_t stream) {
    const float* x      = (const float*)d_in[0];
    const float* style  = (const float*)d_in[1];
    const float* weight = (const float*)d_in[2];
    const float* mod_w  = (const float*)d_in[3];
    const float* mod_b  = (const float*)d_in[4];
    float* out = (float*)d_out;

    float* sbuf = (float*)d_ws;                                     // 16 KB
    unsigned short* wmod = (unsigned short*)((char*)d_ws + 16384);  // 36 MB bf16

    k_modulate  <<<512, 256, 0, stream>>>(style, mod_w, mod_b, sbuf);
    k_demod_pack<<<512, 256, 0, stream>>>(weight, sbuf, wmod);
    k_conv_wmma <<<512, 512, 0, stream>>>(x, wmod, out);
}
